// BidirectionalRNN_83167746719993
// MI455X (gfx1250) — matrix operand order repaired and checked by reading, not compiled, not run on hardware
//
#include <hip/hip_runtime.h>
#include <stdint.h>
#include <math.h>

// ---------------------------------------------------------------------------
// BidirectionalRNN on MI455X (gfx1250): bf16 WMMA (f32 accum) everywhere.
// Shapes: B=32, S=512, D=512, H=512, OUT=256.
// ---------------------------------------------------------------------------

#define Bb 32
#define Ss 512
#define Dd 512
#define Hh 512
#define OUTN 256

typedef __bf16 bf16_t;
typedef __bf16 v16bf __attribute__((ext_vector_type(16)));
typedef float  v8f   __attribute__((ext_vector_type(8)));
typedef uint32_t u32x4 __attribute__((ext_vector_type(4)));

union Frag { v16bf v; u32x4 u[2]; };

__device__ __forceinline__ uint16_t f2bf(float f) {
  uint32_t u = __builtin_bit_cast(uint32_t, f);
  uint32_t r = u + 0x7FFFu + ((u >> 16) & 1u);   // round-to-nearest-even
  return (uint16_t)(r >> 16);
}
__device__ __forceinline__ float bf2f(uint16_t h) {
  uint32_t u = ((uint32_t)h) << 16;
  return __builtin_bit_cast(float, u);
}

// A-fragment (16x32 bf16, row-major source, row stride ldk elements).
// Lane layout (ISA 7.12.2): lane<16 -> K chunks {0..7,16..23}; lane>=16 -> +8.
__device__ __forceinline__ Frag load_a_rm(const uint16_t* __restrict__ base,
                                          int m0, int ldk, int kc, int lane) {
  const int m  = m0 + (lane & 15);
  const int kb = kc + ((lane >> 4) & 1) * 8;
  const uint16_t* p = base + (size_t)m * ldk + kb;
  Frag f;
  f.u[0] = *(const u32x4*)(p);
  f.u[1] = *(const u32x4*)(p + 16);
  return f;
}

// B-fragment from pre-transposed weights WT (N rows x K cols, row-major):
// lane<16 -> K chunks {0..7,16..23}; lane>=16 -> +8; N = lane%16.
__device__ __forceinline__ Frag load_b_T(const uint16_t* __restrict__ baseT,
                                         int n0, int ldk, int kc, int lane) {
  const int n  = n0 + (lane & 15);
  const int kb = kc + ((lane >> 4) & 1) * 8;
  const uint16_t* p = baseT + (size_t)n * ldk + kb;
  Frag f;
  f.u[0] = *(const u32x4*)(p);
  f.u[1] = *(const u32x4*)(p + 16);
  return f;
}

__device__ __forceinline__ v8f wmma_bf16(const Frag& a, const Frag& b, v8f c) {
  return __builtin_amdgcn_wmma_f32_16x16x32_bf16(
      /*neg_a=*/false, a.v, /*neg_b=*/false, b.v,
      /*c_mod=*/(short)0, c, /*reuse_a=*/false, /*reuse_b=*/false);
}

// ---------------------------------------------------------------------------
// Prep kernels
// ---------------------------------------------------------------------------
__global__ void cvt_f32_bf16(const float* __restrict__ in,
                             uint16_t* __restrict__ out, int n) {
  int i = blockIdx.x * blockDim.x + threadIdx.x;
  if (i < n) out[i] = f2bf(in[i]);
}

// out[c][r] = bf16(in[r][c])   in: (R,C) f32, out: (C,R) bf16
__global__ void transpose_f32_bf16(const float* __restrict__ in,
                                   uint16_t* __restrict__ out, int R, int C) {
  int i = blockIdx.x * blockDim.x + threadIdx.x;
  if (i >= R * C) return;
  int c = i / R, r = i - c * R;
  out[(size_t)c * R + r] = f2bf(in[(size_t)r * C + c]);
}

// ---------------------------------------------------------------------------
// Generic WMMA GEMM:  C(M,N) f32 = A(M,K) bf16  @  BT(N,K)^T bf16  + bias(N)
// One wave computes a 32x64 tile (2 A-frags reuse each B-frag; 8 accums).
// permuteSB: row m=(b*S+s) is stored at row s*B+b (scan layout).
// ---------------------------------------------------------------------------
__global__ __launch_bounds__(256) void gemm_bf16_wmma(
    const uint16_t* __restrict__ A, const uint16_t* __restrict__ BT,
    const float* __restrict__ bias, float* __restrict__ C,
    int M, int N, int K, int permuteSB) {
  const int lane = threadIdx.x & 31;
  const int wave = threadIdx.x >> 5;
  const int gw = blockIdx.x * 8 + wave;
  const int ntn = N >> 6;                 // # of 64-wide strips
  const int nmt = (M + 31) >> 5;          // # of 32-row strips
  const int total = nmt * ntn;
  if (gw >= total) return;
  const int mt = gw / ntn;
  const int n64 = gw - mt * ntn;
  const int m0 = mt << 5;
  const int n0 = n64 << 6;

  v8f acc[2][4] = {};
  for (int kc = 0; kc < K; kc += 32) {
    Frag a0 = load_a_rm(A, m0, K, kc, lane);
    Frag a1 = load_a_rm(A, m0 + 16, K, kc, lane);
#pragma unroll
    for (int j = 0; j < 4; ++j) {
      Frag b = load_b_T(BT, n0 + j * 16, K, kc, lane);
      acc[0][j] = wmma_bf16(a0, b, acc[0][j]);
      acc[1][j] = wmma_bf16(a1, b, acc[1][j]);
    }
  }

  const int nlo = lane & 15;
  const int hi = (lane >> 4) & 1;
#pragma unroll
  for (int mi = 0; mi < 2; ++mi) {
#pragma unroll
    for (int j = 0; j < 4; ++j) {
      const int n = n0 + j * 16 + nlo;
      const float bv = bias ? bias[n] : 0.0f;
#pragma unroll
      for (int v = 0; v < 8; ++v) {
        const int m = m0 + mi * 16 + v + hi * 8;
        float val = acc[mi][j][v] + bv;
        size_t row;
        if (permuteSB) {                   // m = b*S + s  ->  row = s*B + b
          const int bb = m >> 9;           // /S (S=512)
          const int ss = m & 511;
          row = (size_t)ss * Bb + bb;
        } else {
          row = (size_t)m;
        }
        C[row * N + n] = val;
      }
    }
  }
}

// ---------------------------------------------------------------------------
// Recurrent scan:  h_{t} = tanh( xw[t] + h_{t-1} @ Wh ),  h in LDS (bf16).
// grid.x = 2 (dir 0 = forward, dir 1 = backward), 512 threads = 16 waves.
// Per step: M=32, N=512, K=512 -> 64 tiles; each wave owns 4 tiles.
// xw[t+1] is prefetched into a double-buffered LDS stage with CDNA5 async
// copies (GLOBAL_LOAD_ASYNC_TO_LDS_B128 / ASYNCcnt), overlapping the WMMA
// K-loop of step t. Each wave copies only its own 16x64 fp32 slice, so a
// per-wave s_wait_asynccnt suffices (no extra barrier).
// ---------------------------------------------------------------------------
__global__ __launch_bounds__(512) void rnn_scan(
    const float* __restrict__ xw_f, const float* __restrict__ xw_b,
    const uint16_t* __restrict__ WhT_f, const uint16_t* __restrict__ WhT_b,
    uint16_t* __restrict__ yout, int ycols,        // null => don't store ys
    float* __restrict__ hfin_f, float* __restrict__ hfin_b,
    uint16_t* __restrict__ seqrep) {               // null for layer 0
  __shared__ __align__(16) uint16_t hbuf[Bb][Hh];        // 32 KB
  __shared__ __align__(16) float xstage[2][16][16][64];  // 128 KB (dbl-buffered)

  const int dir = blockIdx.x;
  const float* __restrict__ xw = dir ? xw_b : xw_f;
  const uint16_t* __restrict__ WhT = dir ? WhT_b : WhT_f;
  float* __restrict__ hfin = dir ? hfin_b : hfin_f;

  for (int i = threadIdx.x; i < Bb * Hh; i += blockDim.x)
    ((uint16_t*)hbuf)[i] = 0;

  const int lane = threadIdx.x & 31;
  const int wave = threadIdx.x >> 5;   // 0..15
  const int mt = wave & 1;             // rows 0-15 or 16-31
  const int ntb = (wave >> 1) << 2;    // first of 4 n-tiles: 0,4,...,28
  const int m0 = mt << 4;
  const int nbase = ntb << 4;          // column base of this wave's 64 cols
  const int nlo = lane & 15;
  const int hi = (lane >> 4) & 1;
  const int a_m = m0 + nlo;
  const int a_koff = hi * 8;

  // Async-copy this wave's 16x64 fp32 slice of xw[t] into xstage[bufi].
  // Lane l covers rows r=(l>>4)+2*i, 4 floats at col (l&15)*4, i=0..7.
  const int pr0 = lane >> 4;
  const int pc4 = (lane & 15) * 4;
  auto prefetch_xw = [&](int t, int bufi) {
    const float* src = xw + ((size_t)t * Bb + m0) * Hh + nbase;
#pragma unroll
    for (int i = 0; i < 8; ++i) {
      const int r = pr0 + 2 * i;
      uint64_t gaddr = (uint64_t)(uintptr_t)(src + (size_t)r * Hh + pc4);
      // Flat->LDS aperture: low 32 bits of a generic LDS pointer are the
      // wave-relative LDS byte offset (ISA 10.2 aperture mapping).
      uint32_t laddr = (uint32_t)(uintptr_t)(&xstage[bufi][wave][r][pc4]);
      asm volatile("global_load_async_to_lds_b128 %0, %1, off"
                   :: "v"(laddr), "v"(gaddr) : "memory");
    }
  };

  // Prologue: stage xw for the first timestep.
  prefetch_xw(dir ? (Ss - 1) : 0, 0);

  for (int step = 0; step < Ss; ++step) {
    const int t = dir ? (Ss - 1 - step) : step;

    // Kick off next step's xw copy; it overlaps the whole WMMA K-loop.
    if (step + 1 < Ss)
      prefetch_xw(dir ? (Ss - 2 - step) : (step + 1), (step + 1) & 1);

    __syncthreads();                   // h from previous step visible

    v8f acc[4] = {};
    for (int kc = 0; kc < Hh; kc += 32) {
      Frag a;
      {
        const uint16_t* p = &hbuf[a_m][kc + a_koff];   // ds_load_b128 x2
        a.u[0] = *(const u32x4*)(p);
        a.u[1] = *(const u32x4*)(p + 16);
      }
#pragma unroll
      for (int j = 0; j < 4; ++j) {
        Frag b = load_b_T(WhT, (ntb + j) * 16, Hh, kc, lane);
        acc[j] = wmma_bf16(a, b, acc[j]);
      }
    }

    // Retire the older async buffer (this step's xw). Async loads complete
    // in order, so <=8 leaves only the just-issued next-step copies pending.
    if (step + 1 < Ss) asm volatile("s_wait_asynccnt 0x8" ::: "memory");
    else               asm volatile("s_wait_asynccnt 0x0" ::: "memory");

    __syncthreads();                   // everyone done reading old h

    const int bufi = step & 1;
#pragma unroll
    for (int j = 0; j < 4; ++j) {
      const int n = (ntb + j) * 16 + nlo;
#pragma unroll
      for (int v = 0; v < 8; ++v) {
        const int r = v + hi * 8;
        const int m = m0 + r;
        const float val = tanhf(acc[j][v] + xstage[bufi][wave][r][j * 16 + nlo]);
        const uint16_t hb = f2bf(val);
        hbuf[m][n] = hb;
        if (yout)
          yout[((size_t)m * Ss + t) * ycols + dir * Hh + n] = hb;
      }
    }
  }

  __syncthreads();
  for (int i = threadIdx.x; i < Bb * Hh; i += blockDim.x) {
    const int b = i >> 9, hh = i & 511;
    hfin[i] = bf2f(hbuf[b][hh]);
    if (seqrep) seqrep[(size_t)b * (2 * Hh) + dir * Hh + hh] = hbuf[b][hh];
  }
}

// ---------------------------------------------------------------------------
// Host orchestration
// ---------------------------------------------------------------------------
extern "C" void kernel_launch(void* const* d_in, const int* in_sizes, int n_in,
                              void* d_out, int out_size, void* d_ws, size_t ws_size,
                              hipStream_t stream) {
  (void)in_sizes; (void)n_in; (void)out_size; (void)ws_size;
  const float* x   = (const float*)d_in[0];
  const float* Wf0 = (const float*)d_in[1];
  const float* bf0 = (const float*)d_in[2];
  const float* Wb0 = (const float*)d_in[3];
  const float* bb0 = (const float*)d_in[4];
  const float* Wf1 = (const float*)d_in[5];
  const float* bf1 = (const float*)d_in[6];
  const float* Wb1 = (const float*)d_in[7];
  const float* bb1 = (const float*)d_in[8];
  const float* Wfc = (const float*)d_in[9];
  const float* bfc = (const float*)d_in[10];
  float* out = (float*)d_out;

  // Workspace carve-up (256B aligned)
  char* ws = (char*)d_ws;
  size_t off = 0;
  auto alloc = [&](size_t bytes) -> char* {
    char* p = ws + off;
    off = (off + bytes + 255) & ~(size_t)255;
    return p;
  };
  uint16_t* xbf    = (uint16_t*)alloc((size_t)Bb * Ss * Dd * 2);       // 16 MB
  uint16_t* WxT0f  = (uint16_t*)alloc((size_t)Hh * Dd * 2);
  uint16_t* WhT0f  = (uint16_t*)alloc((size_t)Hh * Hh * 2);
  uint16_t* WxT0b  = (uint16_t*)alloc((size_t)Hh * Dd * 2);
  uint16_t* WhT0b  = (uint16_t*)alloc((size_t)Hh * Hh * 2);
  uint16_t* WxT1f  = (uint16_t*)alloc((size_t)Hh * 2 * Hh * 2);
  uint16_t* WhT1f  = (uint16_t*)alloc((size_t)Hh * Hh * 2);
  uint16_t* WxT1b  = (uint16_t*)alloc((size_t)Hh * 2 * Hh * 2);
  uint16_t* WhT1b  = (uint16_t*)alloc((size_t)Hh * Hh * 2);
  uint16_t* WfcT   = (uint16_t*)alloc((size_t)OUTN * 2 * Hh * 2);
  float*    xw_f   = (float*)alloc((size_t)Ss * Bb * Hh * 4);          // 32 MB (reused L0/L1)
  float*    xw_b   = (float*)alloc((size_t)Ss * Bb * Hh * 4);          // 32 MB
  uint16_t* y0     = (uint16_t*)alloc((size_t)Bb * Ss * 2 * Hh * 2);   // 32 MB
  uint16_t* seqrep = (uint16_t*)alloc((size_t)Bb * 2 * Hh * 2);

  // 1) x -> bf16
  {
    int n = Bb * Ss * Dd;
    cvt_f32_bf16<<<(n + 255) / 256, 256, 0, stream>>>(x, xbf, n);
  }
  // 2) weight transposes (+bf16). W[:Din] = Wx, W[Din:] = Wh.
  auto tr = [&](const float* in, uint16_t* o, int R, int C) {
    int n = R * C;
    transpose_f32_bf16<<<(n + 255) / 256, 256, 0, stream>>>(in, o, R, C);
  };
  tr(Wf0,               WxT0f, Dd, Hh);
  tr(Wf0 + Dd * Hh,     WhT0f, Hh, Hh);
  tr(Wb0,               WxT0b, Dd, Hh);
  tr(Wb0 + Dd * Hh,     WhT0b, Hh, Hh);
  tr(Wf1,               WxT1f, 2 * Hh, Hh);
  tr(Wf1 + 2 * Hh * Hh, WhT1f, Hh, Hh);
  tr(Wb1,               WxT1b, 2 * Hh, Hh);
  tr(Wb1 + 2 * Hh * Hh, WhT1b, Hh, Hh);
  tr(Wfc,               WfcT, 2 * Hh, OUTN);

  auto gemm = [&](const uint16_t* A, const uint16_t* BT, const float* bias,
                  float* C, int M, int N, int K, int permute) {
    int waves = ((M + 31) / 32) * (N / 64);
    int blocks = (waves + 7) / 8;
    gemm_bf16_wmma<<<blocks, 256, 0, stream>>>(A, BT, bias, C, M, N, K, permute);
  };

  // Output layout: logits(32*256) | final_hf(2*32*512) | final_hb(2*32*512)
  float* logits = out;
  float* hf0 = out + Bb * OUTN;
  float* hf1 = hf0 + Bb * Hh;
  float* hb0 = hf1 + Bb * Hh;
  float* hb1 = hb0 + Bb * Hh;

  // 3) layer-0 input projections: xw = x @ Wx + b  (stored as (S,B,H))
  gemm(xbf, WxT0f, bf0, xw_f, Bb * Ss, Hh, Dd, /*permute=*/1);
  gemm(xbf, WxT0b, bb0, xw_b, Bb * Ss, Hh, Dd, /*permute=*/1);

  // 4) layer-0 scans -> y0 (B,S,2H) bf16, final states fp32
  rnn_scan<<<2, 512, 0, stream>>>(xw_f, xw_b, WhT0f, WhT0b,
                                  y0, 2 * Hh, hf0, hb0, nullptr);

  // 5) layer-1 input projections: xw1 = y0 @ Wx1 + b1 (K = 2H)
  gemm(y0, WxT1f, bf1, xw_f, Bb * Ss, Hh, 2 * Hh, /*permute=*/1);
  gemm(y0, WxT1b, bb1, xw_b, Bb * Ss, Hh, 2 * Hh, /*permute=*/1);

  // 6) layer-1 scans -> final states + seq_repr (no ys needed)
  rnn_scan<<<2, 512, 0, stream>>>(xw_f, xw_b, WhT1f, WhT1b,
                                  nullptr, 0, hf1, hb1, seqrep);

  // 7) logits = seq_repr @ Wfc + bfc   (M=32, N=256, K=1024)
  gemm(seqrep, WfcT, bfc, logits, Bb, OUTN, 2 * Hh, /*permute=*/0);
}
